// Attention_40759239639180
// MI455X (gfx1250) — matrix operand order repaired and checked by reading, not compiled, not run on hardware
//
#include <hip/hip_runtime.h>
#include <cstdint>

// ---------------------------------------------------------------------------
// Additive (Bahdanau) attention for MI455X (gfx1250, wave32, WMMA).
// Pipeline:
//   1) kx = fold(k@Wk + bk), qx = fold(q@Wq + bq)      [WMMA GEMM, MODE 0]
//   2) ks = kx . w[:128], qs = qx . w[128:]            [wave row-dot]
//   3) score = softmax(tanh(qs[:,None]+ks[None,:]))    [fused streaming kernel]
//   4) ctx = score @ kx  (batched over hb=64)          [WMMA GEMM, MODE 1]
//   5) out = unfold(ctx) @ Wp + bp                     [WMMA GEMM, MODE 2]
// GEMM inner loop feeds v_wmma_f32_16x16x32_bf16 from fragment-ready bf16 LDS
// tiles (one f32->bf16 conversion per element at staging, wide b128 DS ops).
// ---------------------------------------------------------------------------

typedef __attribute__((ext_vector_type(16))) __bf16 v16bf;
typedef __attribute__((ext_vector_type(8)))  float  v8f;

#define MBATCH 8
#define SLEN   1024
#define DIN    1024
#define NHEAD  8
#define HID    128
#define HB     (NHEAD * MBATCH)   // 64

#define BM 128
#define BN 128
#define BK 32

// MODE 0: A plain row-major, C scattered into folded-head layout, bias added.
// MODE 1: batched (blockIdx.z): A, B, C all plain row-major per batch, no bias.
// MODE 2: A gathered from folded-head layout, C plain row-major, bias added.
template <int MODE>
__global__ __launch_bounds__(256) void gemm_wmma_kernel(
    const float* __restrict__ A, const float* __restrict__ B,
    const float* __restrict__ bias, float* __restrict__ C,
    int M, int N, int K, int lda, int ldb, int ldc)
{
    // Fragment-ready bf16 tiles:
    //  Asw[(row*2 + g)*16 + i]  : g = lane>>4, values K = g*16+{0..15}
    //  Bsw[(t*32 + lane)*16 + i]: lane = col + 16*(K>>4), values K = (lane>>4)*16+{0..15}
    __shared__ __align__(16) __bf16 Asw[BM * BK];   // 8 KB
    __shared__ __align__(16) __bf16 Bsw[BK * BN];   // 8 KB

    const int tid  = threadIdx.x;
    const int lane = tid & 31;
    const int wave = tid >> 5;                 // 0..7, wave owns rows [wave*16, wave*16+16)
    const int gm   = blockIdx.y * BM;
    const int gn   = blockIdx.x * BN;

    const float* Ab = A;
    const float* Bb = B;
    float*       Cb = C;
    if (MODE == 1) {
        const size_t b = blockIdx.z;
        Ab += b * (size_t)M * (size_t)lda;
        Bb += b * (size_t)K * (size_t)ldb;
        Cb += b * (size_t)M * (size_t)ldc;
    }

    v8f acc[8] = {};

    // ---- staging assignments (256 threads, 16 elements each, both tiles) ----
    const int ar = tid >> 1;                   // A: row 0..127
    const int ag = tid & 1;                    // A: lane-group (K sub-block)
    const int bn = tid & 127;                  // B: column 0..127
    const int bg = tid >> 7;                   // B: K half 0..1
    __bf16* adst = &Asw[(ar * 2 + ag) * 16];
    __bf16* bdst = &Bsw[((bn >> 4) * 32 + (bg * 16 + (bn & 15))) * 16];

    // ---- fragment read addresses ----
    const __bf16* afr = &Asw[((wave * 16 + (lane & 15)) * 2 + (lane >> 4)) * 16];

    for (int k0 = 0; k0 < K; k0 += BK) {
        __syncthreads();   // previous iteration's fragment reads done before LDS overwrite

        // ---- stage A: 2x8 contiguous f32 -> 16 bf16 in fragment order ----
        {
            const int grow = gm + ar;
            const int c0   = k0 + ag * 16;      // K offsets c0..c0+7 and c0+8..c0+15
            const float* s0;
            const float* s1;
            if (MODE == 2) {
                const int mb = grow >> 10, seq = grow & 1023;
                const int h0 = c0 >> 7,        hc0 = c0 & 127;
                const int h1 = (c0 + 8) >> 7, hc1 = (c0 + 8) & 127;
                const size_t rbase = (size_t)seq * HID;
                s0 = Ab + ((size_t)(h0 * MBATCH + mb) * SLEN) * HID + rbase + hc0;
                s1 = Ab + ((size_t)(h1 * MBATCH + mb) * SLEN) * HID + rbase + hc1;
            } else {
                s0 = Ab + (size_t)grow * (size_t)lda + c0;
                s1 = s0 + 8;
            }
            const float4 a0 = *(const float4*)(s0);
            const float4 a1 = *(const float4*)(s0 + 4);
            const float4 a2 = *(const float4*)(s1);
            const float4 a3 = *(const float4*)(s1 + 4);
            v16bf p;
            p[0]  = (__bf16)a0.x; p[1]  = (__bf16)a0.y; p[2]  = (__bf16)a0.z; p[3]  = (__bf16)a0.w;
            p[4]  = (__bf16)a1.x; p[5]  = (__bf16)a1.y; p[6]  = (__bf16)a1.z; p[7]  = (__bf16)a1.w;
            p[8]  = (__bf16)a2.x; p[9]  = (__bf16)a2.y; p[10] = (__bf16)a2.z; p[11] = (__bf16)a2.w;
            p[12] = (__bf16)a3.x; p[13] = (__bf16)a3.y; p[14] = (__bf16)a3.z; p[15] = (__bf16)a3.w;
            *(v16bf*)adst = p;
        }
        // ---- stage B: 16 coalesced column reads -> 16 bf16 in fragment order ----
        {
            const float* s = Bb + (size_t)(k0 + bg * 16) * (size_t)ldb + gn + bn;
            v16bf p;
            #pragma unroll
            for (int i = 0; i < 16; ++i) p[i] = (__bf16)s[(size_t)i * (size_t)ldb];
            *(v16bf*)bdst = p;
            if (k0 + BK < K)   // hint next B tile -> global_prefetch_b8
                __builtin_prefetch(s + (size_t)BK * (size_t)ldb, 0, 1);
        }
        __syncthreads();

        // ---- inner compute: 1 A-frag + 8 (B-frag + WMMA) ----
        const v16bf af = *(const v16bf*)afr;
        #pragma unroll
        for (int t = 0; t < 8; ++t) {
            const v16bf bfr = *(const v16bf*)&Bsw[(t * 32 + lane) * 16];
            acc[t] = __builtin_amdgcn_wmma_f32_16x16x32_bf16(
                         false, af, false, bfr, (short)0, acc[t], false, false);
        }
    }

    // ---- epilogue: C 16x16 f32 layout = VGPR v: M = v + 8*(lane>=16), N = lane&15 ----
    const int mrow = wave * 16 + ((lane < 16) ? 0 : 8);
    const int ncol = lane & 15;
    #pragma unroll
    for (int t = 0; t < 8; ++t) {
        const int col = gn + t * 16 + ncol;
        const float bv = (MODE != 1) ? bias[col] : 0.0f;
        #pragma unroll
        for (int v = 0; v < 8; ++v) {
            const int row = gm + mrow + v;
            const float val = acc[t][v] + bv;
            if (MODE == 0) {
                const int mb = row >> 10, seq = row & 1023;
                const int h  = col >> 7,  hc  = col & 127;
                Cb[((size_t)((h * MBATCH + mb) * SLEN + seq)) * HID + hc] = val;
            } else {
                Cb[(size_t)row * (size_t)ldc + col] = val;
            }
        }
    }
}

// rows of 128 elements dotted with a 128-vector; one wave per row
__global__ __launch_bounds__(256) void rowdot_kernel(
    const float* __restrict__ X, const float* __restrict__ w,
    float* __restrict__ out)
{
    const int lane = threadIdx.x & 31;
    const int wave = threadIdx.x >> 5;
    const int row  = blockIdx.x * 8 + wave;
    const float* xr = X + (size_t)row * HID;
    float s = 0.0f;
    #pragma unroll
    for (int i = lane; i < HID; i += 32) s += xr[i] * w[i];
    #pragma unroll
    for (int off = 16; off; off >>= 1) s += __shfl_xor(s, off, 32);
    if (lane == 0) out[row] = s;
}

// score[hb, q, :] = softmax_j( tanh(qs[hb,q] + ks[hb,j]) ), one block per (hb,q)
__global__ __launch_bounds__(256) void softmax_kernel(
    const float* __restrict__ ks, const float* __restrict__ qs,
    float* __restrict__ score)
{
    const int hb   = blockIdx.y;
    const int qrow = blockIdx.x;
    const int tid  = threadIdx.x;
    const int lane = tid & 31;
    const int wave = tid >> 5;

    const float qv = qs[hb * SLEN + qrow];
    const float* krow = ks + (size_t)hb * SLEN;

    float v[4];
    float mx = -1e30f;
    #pragma unroll
    for (int i = 0; i < 4; ++i) {
        v[i] = tanhf(qv + krow[tid + i * 256]);
        mx = fmaxf(mx, v[i]);
    }
    #pragma unroll
    for (int off = 16; off; off >>= 1) mx = fmaxf(mx, __shfl_xor(mx, off, 32));

    __shared__ float smax[8], ssum[8];
    if (lane == 0) smax[wave] = mx;
    __syncthreads();
    float rowmax = smax[0];
    #pragma unroll
    for (int i = 1; i < 8; ++i) rowmax = fmaxf(rowmax, smax[i]);

    float s = 0.0f;
    #pragma unroll
    for (int i = 0; i < 4; ++i) { v[i] = __expf(v[i] - rowmax); s += v[i]; }
    #pragma unroll
    for (int off = 16; off; off >>= 1) s += __shfl_xor(s, off, 32);
    if (lane == 0) ssum[wave] = s;
    __syncthreads();
    float tot = ssum[0];
    #pragma unroll
    for (int i = 1; i < 8; ++i) tot += ssum[i];
    const float inv = 1.0f / tot;

    float* orow = score + ((size_t)hb * SLEN + qrow) * SLEN;
    #pragma unroll
    for (int i = 0; i < 4; ++i) orow[tid + i * 256] = v[i] * inv;
}

extern "C" void kernel_launch(void* const* d_in, const int* in_sizes, int n_in,
                              void* d_out, int out_size, void* d_ws, size_t ws_size,
                              hipStream_t stream)
{
    (void)in_sizes; (void)n_in; (void)out_size; (void)ws_size;
    const float* k  = (const float*)d_in[0];
    const float* q  = (const float*)d_in[1];
    const float* Wk = (const float*)d_in[2];
    const float* bk = (const float*)d_in[3];
    const float* Wq = (const float*)d_in[4];
    const float* bq = (const float*)d_in[5];
    const float* w  = (const float*)d_in[6];
    const float* Wp = (const float*)d_in[7];
    const float* bp = (const float*)d_in[8];

    float* out   = (float*)d_out;                                    // (8,1024,1024)
    float* score = out + (size_t)MBATCH * SLEN * DIN;                // (64,1024,1024)

    const size_t NFOLD = (size_t)HB * SLEN * HID;                    // 8.38M floats
    float* kx  = (float*)d_ws;
    float* qx  = kx + NFOLD;
    float* ctx = qx + NFOLD;
    float* ks  = ctx + NFOLD;                                        // 64*1024
    float* qs  = ks + (size_t)HB * SLEN;

    const int nrow = MBATCH * SLEN;                                  // 8192

    // 1) projections (folded-head output)
    gemm_wmma_kernel<0><<<dim3(DIN / BN, nrow / BM, 1), 256, 0, stream>>>(
        k, Wk, bk, kx, nrow, DIN, DIN, DIN, DIN, DIN);
    gemm_wmma_kernel<0><<<dim3(DIN / BN, nrow / BM, 1), 256, 0, stream>>>(
        q, Wq, bq, qx, nrow, DIN, DIN, DIN, DIN, DIN);

    // 2) per-row score vectors
    rowdot_kernel<<<dim3(HB * SLEN / 8), 256, 0, stream>>>(kx, w, ks);
    rowdot_kernel<<<dim3(HB * SLEN / 8), 256, 0, stream>>>(qx, w + HID, qs);

    // 3) fused tanh + softmax, streamed straight into d_out's score region
    softmax_kernel<<<dim3(SLEN, HB), 256, 0, stream>>>(ks, qs, score);

    // 4) ctx = score @ kx, batched over hb (N = HID = 128, single column tile)
    gemm_wmma_kernel<1><<<dim3(1, SLEN / BM, HB), 256, 0, stream>>>(
        score, kx, nullptr, ctx, SLEN, HID, SLEN, SLEN, HID, HID);

    // 5) out = unfold(ctx) @ Wp + bp
    gemm_wmma_kernel<2><<<dim3(DIN / BN, nrow / BM, 1), 256, 0, stream>>>(
        ctx, Wp, bp, out, nrow, DIN, DIN, DIN, DIN, DIN);
}
